// LSTMBranch_46686294508070
// MI455X (gfx1250) — matrix operand order repaired and checked by reading, not compiled, not run on hardware
//
#include <hip/hip_runtime.h>
#include <hip/hip_bf16.h>
#include <math.h>

// Problem dims (from reference)
#define DIN  64
#define DEV  32
#define HID  256
#define G4   1024   // 4*HID
#define BSZ  128
#define TSEQ 1024
#define P1   128    // HID/2

typedef __attribute__((ext_vector_type(16))) _Float16 v16h;
typedef __attribute__((ext_vector_type(8)))  float    v8f;

// ---------------------------------------------------------------- WMMA helpers

__device__ __forceinline__ v8f wmma16(v16h a, v16h b, v8f c) {
  // v_wmma_f32_16x16x32_f16 : D = A(16x32 f16) * B(32x16 f16) + C(16x16 f32)
  return __builtin_amdgcn_wmma_f32_16x16x32_f16(
      /*neg_a=*/false, a, /*neg_b=*/false, b,
      /*c_mod=*/(short)0, c, /*reuse_a=*/false, /*reuse_b=*/false);
}

// A-matrix fragment (16x32 slice at kbase) from row-major f16 [rows][ld].
// ISA layout: lanes 0-15 -> M=lane, VGPR v: K=2v (v<4) / 2v+8 (v>=4); lanes 16-31: +8.
__device__ __forceinline__ v16h load_a16(const _Float16* __restrict__ p, int ld,
                                         int row, int kbase, int g) {
  v16h f;
#pragma unroll
  for (int v = 0; v < 8; ++v) {
    int k0 = kbase + 2 * v + ((v & 4) ? 8 : 0) + 8 * g;
    f[2 * v]     = p[row * ld + k0];
    f[2 * v + 1] = p[row * ld + k0 + 1];
  }
  return f;
}

// Same A layout but from an f32 source (features re-read by predictor).
__device__ __forceinline__ v16h load_a32(const float* __restrict__ p, int ld,
                                         int row, int kbase, int g) {
  v16h f;
#pragma unroll
  for (int v = 0; v < 8; ++v) {
    int k0 = kbase + 2 * v + ((v & 4) ? 8 : 0) + 8 * g;
    f[2 * v]     = (_Float16)p[row * ld + k0];
    f[2 * v + 1] = (_Float16)p[row * ld + k0 + 1];
  }
  return f;
}

// B-matrix fragment (32x16 slice at kbase) from TRANSPOSED weights [N][K] f16.
// Layout matches A: lane = N, VGPR v: K=2v (v<4) / 2v+8 (v>=4); lanes 16-31: +8.
__device__ __forceinline__ v16h load_b16(const _Float16* __restrict__ p, int ld,
                                         int nrow, int kbase, int g) {
  v16h f;
#pragma unroll
  for (int v = 0; v < 8; ++v) {
    int k0 = kbase + 2 * v + ((v & 4) ? 8 : 0) + 8 * g;
    f[2 * v]     = p[nrow * ld + k0];
    f[2 * v + 1] = p[nrow * ld + k0 + 1];
  }
  return f;
}

// ---- fast transcendentals on the serial critical path (v_exp_f32 / v_rcp_f32)
#define LOG2E 1.4426950408889634f
__device__ __forceinline__ float fexp(float z)  { return __builtin_amdgcn_exp2f(z * LOG2E); }
__device__ __forceinline__ float fsigm(float z) { return __builtin_amdgcn_rcpf(1.0f + fexp(-z)); }
__device__ __forceinline__ float ftanh(float z) { return 2.0f * fsigm(2.0f * z) - 1.0f; }

// ---------------------------------------------------------------- prep kernels

// Wpi_t[n][k] = sum_h W_proj[k][h] * W_ih[h][n]   (fold input proj into LSTM weights)
__global__ void k_fold_wpi(const float* __restrict__ Wproj, const float* __restrict__ Wih,
                           _Float16* __restrict__ Wpi_t) {
  int i = blockIdx.x * 256 + threadIdx.x;
  if (i >= G4 * DIN) return;
  int n = i / DIN, k = i - n * DIN;
  float s = 0.f;
  for (int h = 0; h < HID; ++h) s += Wproj[k * HID + h] * Wih[h * G4 + n];
  Wpi_t[n * DIN + k] = (_Float16)s;
}

// b_g[n] = b_lstm[n] + sum_h b_proj[h] * W_ih[h][n]
__global__ void k_fold_bias(const float* __restrict__ bproj, const float* __restrict__ Wih,
                            const float* __restrict__ blstm, float* __restrict__ bg) {
  int n = blockIdx.x * 256 + threadIdx.x;
  if (n >= G4) return;
  float s = blstm[n];
  for (int h = 0; h < HID; ++h) s += bproj[h] * Wih[h * G4 + n];
  bg[n] = s;
}

// transpose-convert: src f32 [K][N] row-major -> dst f16 [N][K]
__global__ void k_t16(const float* __restrict__ src, _Float16* __restrict__ dst,
                      int K, int N) {
  int i = blockIdx.x * 256 + threadIdx.x;
  if (i >= K * N) return;
  int k = i / N, n = i - k * N;
  dst[n * K + k] = (_Float16)src[i];
}

// ---------------------------------------------------------------- recurrence

#define HLD (HID + 8)
#define XLD (DIN + 8)
#define ELD (DEV + 8)

__global__ __launch_bounds__(256)
void k_recur(const float* __restrict__ x, const float* __restrict__ ev,
             const float* __restrict__ td,
             const _Float16* __restrict__ Whh,  // [1024][256]
             const _Float16* __restrict__ Wpi,  // [1024][64]
             const _Float16* __restrict__ Whg,  // [256][256]
             const _Float16* __restrict__ Wmk,  // [256][64]
             const _Float16* __restrict__ We,   // [256][32]
             const float* __restrict__ bg, const float* __restrict__ bhg,
             const float* __restrict__ bmk, const float* __restrict__ be,
             const float* __restrict__ decay, float* __restrict__ feat) {
  __shared__ _Float16 hbuf[16 * HLD];
  __shared__ _Float16 xbuf[16 * XLD];
  __shared__ _Float16 ebuf[16 * ELD];
  __shared__ float    dtbuf[16];   // holds exp(-softplus(decay)*dt) per row

  const int tid  = threadIdx.x;
  const int wave = tid >> 5;
  const int lane = tid & 31;
  const int l15  = lane & 15;
  const int g    = lane >> 4;
  const int b0   = blockIdx.x * 16;

  const float sp = log1pf(expf(decay[0]));  // softplus(decay), uniform

  // ---- loop-invariant per-lane biases (hoisted out of the t-loop)
  float vbgi[2], vbgf[2], vbgg[2], vbgo[2], vbhg[2], vbmk[2], vbe[2];
#pragma unroll
  for (int tt = 0; tt < 2; ++tt) {
    const int n = wave * 32 + tt * 16 + l15;
    vbgi[tt] = bg[n];        vbgf[tt] = bg[256 + n];
    vbgg[tt] = bg[512 + n];  vbgo[tt] = bg[768 + n];
    vbhg[tt] = bhg[n];       vbmk[tt] = bmk[n];       vbe[tt] = be[n];
  }

  // ---- init: h0 = 0, stage step-0 inputs
  for (int i = tid; i < 16 * HLD; i += 256) hbuf[i] = (_Float16)0.0f;
#pragma unroll
  for (int j = 0; j < 4; ++j) {
    int i = tid + 256 * j, m = i >> 6, k = i & 63;
    xbuf[m * XLD + k] = (_Float16)x[((size_t)(b0 + m) * TSEQ + 0) * DIN + k];
  }
#pragma unroll
  for (int j = 0; j < 2; ++j) {
    int i = tid + 256 * j, m = i >> 5, k = i & 31;
    ebuf[m * ELD + k] = (_Float16)ev[((size_t)(b0 + m) * TSEQ + 0) * DEV + k];
  }
  if (tid < 16) dtbuf[tid] = fexp(-sp * td[(size_t)(b0 + tid) * TSEQ + 0]);
  __syncthreads();

  v8f c_st[2];
  const v8f vzero = {0.f, 0.f, 0.f, 0.f, 0.f, 0.f, 0.f, 0.f};
  c_st[0] = vzero; c_st[1] = vzero;

#pragma unroll 1
  for (int t = 0; t < TSEQ; ++t) {
    // Defeat LICM of the (t-invariant) weight loads without losing the global
    // address space: an opaque scalar row offset (always 0, but unknowable to
    // the compiler) makes every B-fragment load loop-variant, so they stream
    // from L2 each step as global_load_b128 (LOADcnt only) instead of being
    // hoisted+spilled to scratch or demoted to flat_load (LOADcnt+DScnt).
    int zoff = 0;
    asm volatile("" : "+s"(zoff));

    // ---- phase 1a: issue next-step global loads into registers (overlap w/ WMMA)
    const int tn = (t + 1 < TSEQ) ? t + 1 : t;
    float stx[4], ste[2], strdt = 0.f;
#pragma unroll
    for (int j = 0; j < 4; ++j) {
      int i = tid + 256 * j, m = i >> 6, k = i & 63;
      stx[j] = x[((size_t)(b0 + m) * TSEQ + tn) * DIN + k];
    }
#pragma unroll
    for (int j = 0; j < 2; ++j) {
      int i = tid + 256 * j, m = i >> 5, k = i & 31;
      ste[j] = ev[((size_t)(b0 + m) * TSEQ + tn) * DEV + k];
    }
    if (tid < 16) strdt = td[(size_t)(b0 + tid) * TSEQ + tn];

    // snapshot decay factors for step t (dtbuf gets overwritten in phase 2)
    float edk[8];
#pragma unroll
    for (int r = 0; r < 8; ++r) edk[r] = dtbuf[g * 8 + r];

    // ---- phase 1b: A fragments, hoisted once per step (shared by tt=0/1)
    v16h ah[8], axf[2], aef;
#pragma unroll
    for (int kk = 0; kk < 8; ++kk) ah[kk] = load_a16(hbuf, HLD, l15, kk * 32, g);
#pragma unroll
    for (int kk = 0; kk < 2; ++kk) axf[kk] = load_a16(xbuf, XLD, l15, kk * 32, g);
    aef = load_a16(ebuf, ELD, l15, 0, g);

    v8f ai[2], af[2], ag[2], ao[2], agt[2], amk[2], ae[2];
#pragma unroll
    for (int tt = 0; tt < 2; ++tt) {
      ai[tt] = vzero; af[tt] = vzero; ag[tt] = vzero; ao[tt] = vzero;
      agt[tt] = vzero; amk[tt] = vzero; ae[tt] = vzero;
    }

    // ---- phase 1c: 102 WMMAs per wave (weights stream from L2)
#pragma unroll
    for (int tt = 0; tt < 2; ++tt) {
      const int nr = wave * 32 + tt * 16 + l15 + zoff;
      for (int kk = 0; kk < 8; ++kk) {
        ai[tt]  = wmma16(ah[kk], load_b16(Whh, HID, nr,       kk * 32, g), ai[tt]);
        af[tt]  = wmma16(ah[kk], load_b16(Whh, HID, 256 + nr, kk * 32, g), af[tt]);
        ag[tt]  = wmma16(ah[kk], load_b16(Whh, HID, 512 + nr, kk * 32, g), ag[tt]);
        ao[tt]  = wmma16(ah[kk], load_b16(Whh, HID, 768 + nr, kk * 32, g), ao[tt]);
        agt[tt] = wmma16(ah[kk], load_b16(Whg, HID, nr,       kk * 32, g), agt[tt]);
      }
      for (int kk = 0; kk < 2; ++kk) {
        ai[tt]  = wmma16(axf[kk], load_b16(Wpi, DIN, nr,       kk * 32, g), ai[tt]);
        af[tt]  = wmma16(axf[kk], load_b16(Wpi, DIN, 256 + nr, kk * 32, g), af[tt]);
        ag[tt]  = wmma16(axf[kk], load_b16(Wpi, DIN, 512 + nr, kk * 32, g), ag[tt]);
        ao[tt]  = wmma16(axf[kk], load_b16(Wpi, DIN, 768 + nr, kk * 32, g), ao[tt]);
        amk[tt] = wmma16(axf[kk], load_b16(Wmk, DIN, nr,       kk * 32, g), amk[tt]);
      }
      ae[tt] = wmma16(aef, load_b16(We, DEV, nr, 0, g), ae[tt]);
    }
    __syncthreads();  // all LDS reads for step t complete

    // ---- phase 2a: elementwise LSTM combine (all within this wave's registers)
#pragma unroll
    for (int tt = 0; tt < 2; ++tt) {
      const int n = wave * 32 + tt * 16 + l15;
#pragma unroll
      for (int r = 0; r < 8; ++r) {
        const int m = g * 8 + r;
        float zi = ai[tt][r] + vbgi[tt], zf = af[tt][r] + vbgf[tt];
        float zg = ag[tt][r] + vbgg[tt], zo = ao[tt][r] + vbgo[tt];
        float e    = ftanh(ae[tt][r] + vbe[tt]);
        float gate = fsigm(agt[tt][r] + vbhg[tt]);
        float imp  = e * gate * edk[r];
        float mk   = fsigm(amk[tt][r] + vbmk[tt]);
        float c = c_st[tt][r];
        c = fsigm(zf) * c + fsigm(zi) * ftanh(zg) + mk * ftanh(imp);
        c_st[tt][r] = c;
        float h = fsigm(zo) * ftanh(c);
        hbuf[m * HLD + n] = (_Float16)h;
        feat[((size_t)(b0 + m) * TSEQ + t) * HID + n] = h;
      }
    }

    // ---- phase 2b: commit staged t+1 inputs to LDS
#pragma unroll
    for (int j = 0; j < 4; ++j) {
      int i = tid + 256 * j, m = i >> 6, k = i & 63;
      xbuf[m * XLD + k] = (_Float16)stx[j];
    }
#pragma unroll
    for (int j = 0; j < 2; ++j) {
      int i = tid + 256 * j, m = i >> 5, k = i & 31;
      ebuf[m * ELD + k] = (_Float16)ste[j];
    }
    if (tid < 16) dtbuf[tid] = fexp(-sp * strdt);
    __syncthreads();
  }
}

// ---------------------------------------------------------------- predictor

__global__ __launch_bounds__(256)
void k_pred(const float* __restrict__ feat, const _Float16* __restrict__ Wp1,
            const float* __restrict__ bp1, const float* __restrict__ wp2,
            const float* __restrict__ bp2, float* __restrict__ preds) {
  const int wave = threadIdx.x >> 5, lane = threadIdx.x & 31;
  const int l15 = lane & 15, g = lane >> 4;
  const int rt = blockIdx.x * 8 + wave;   // 16-row tile of the B*T rows
  const int r0 = rt * 16;

  v16h a8[8];
#pragma unroll
  for (int kk = 0; kk < 8; ++kk) a8[kk] = load_a32(feat, HID, r0 + l15, kk * 32, g);

  const v8f vzero = {0.f, 0.f, 0.f, 0.f, 0.f, 0.f, 0.f, 0.f};
  v8f acc[8];
#pragma unroll
  for (int nt = 0; nt < 8; ++nt) {
    acc[nt] = vzero;
    for (int kk = 0; kk < 8; ++kk)
      acc[nt] = wmma16(a8[kk], load_b16(Wp1, HID, nt * 16 + l15, kk * 32, g), acc[nt]);
  }

  float sum[8] = {0.f, 0.f, 0.f, 0.f, 0.f, 0.f, 0.f, 0.f};
#pragma unroll
  for (int nt = 0; nt < 8; ++nt) {
    const int n = nt * 16 + l15;
    const float bb = bp1[n], w2 = wp2[n];
#pragma unroll
    for (int r = 0; r < 8; ++r) {
      float z = acc[nt][r] + bb;
      float ge = 0.5f * z * (1.0f + erff(z * 0.70710678118654752f));  // exact GELU
      sum[r] += ge * w2;
    }
  }
  // reduce over 16 lanes within each half-wave (lanes 16-31 hold different M rows)
#pragma unroll
  for (int r = 0; r < 8; ++r) {
    float s = sum[r];
    s += __shfl_xor(s, 1, 16);
    s += __shfl_xor(s, 2, 16);
    s += __shfl_xor(s, 4, 16);
    s += __shfl_xor(s, 8, 16);
    if (l15 == 0) preds[r0 + g * 8 + r] = s + bp2[0];
  }
}

// ---------------------------------------------------------------- tmdo (d1)

__global__ void k_tmdo(const float* __restrict__ x, float* __restrict__ d1) {
  size_t i = (size_t)blockIdx.x * 256 + threadIdx.x;
  if (i >= (size_t)BSZ * TSEQ * DIN) return;
  size_t bt = i / DIN;
  int t = (int)(bt % TSEQ);
  d1[i] = (t == 0) ? 0.0f : x[i] - x[i - DIN];
}

// ---------------------------------------------------------------- launch

extern "C" void kernel_launch(void* const* d_in, const int* in_sizes, int n_in,
                              void* d_out, int out_size, void* d_ws, size_t ws_size,
                              hipStream_t stream) {
  (void)in_sizes; (void)n_in; (void)out_size; (void)ws_size;
  const float* x      = (const float*)d_in[0];
  const float* events = (const float*)d_in[1];
  const float* td     = (const float*)d_in[2];
  const float* Wproj  = (const float*)d_in[3];
  const float* bproj  = (const float*)d_in[4];
  const float* Wih    = (const float*)d_in[5];
  const float* Whh_f  = (const float*)d_in[6];
  const float* blstm  = (const float*)d_in[7];
  const float* Wmk_f  = (const float*)d_in[8];
  const float* bmk    = (const float*)d_in[9];
  const float* We_f   = (const float*)d_in[10];
  const float* be     = (const float*)d_in[11];
  const float* Whg_f  = (const float*)d_in[12];
  const float* bhg    = (const float*)d_in[13];
  const float* decay  = (const float*)d_in[14];
  const float* Wp1_f  = (const float*)d_in[15];
  const float* bp1    = (const float*)d_in[16];
  const float* Wp2    = (const float*)d_in[17];
  const float* bp2    = (const float*)d_in[18];

  // Output layout: preds [B*T], features [B*T*H], tmdo d1 [B*T*DIN]
  float* preds = (float*)d_out;
  float* feat  = preds + (size_t)BSZ * TSEQ;
  float* d1    = feat + (size_t)BSZ * TSEQ * HID;

  // Workspace: f16 transposed weight copies (+ folded bias), ~900 KB total
  char* wsb = (char*)d_ws;
  _Float16* Whh_t = (_Float16*)(wsb + 0);        // [1024][256]  512 KB
  _Float16* Wpi_t = (_Float16*)(wsb + 524288);   // [1024][64]   128 KB
  _Float16* Whg_t = (_Float16*)(wsb + 655360);   // [256][256]   128 KB
  _Float16* Wmk_t = (_Float16*)(wsb + 786432);   // [256][64]     32 KB
  _Float16* We_t  = (_Float16*)(wsb + 819200);   // [256][32]     16 KB
  _Float16* Wp1_t = (_Float16*)(wsb + 835584);   // [128][256]    64 KB
  float*    bg    = (float*)   (wsb + 901120);   // [1024]         4 KB

  // weight prep (tiny)
  k_fold_wpi <<<(G4 * DIN + 255) / 256, 256, 0, stream>>>(Wproj, Wih, Wpi_t);
  k_fold_bias<<<(G4 + 255) / 256,       256, 0, stream>>>(bproj, Wih, blstm, bg);
  k_t16<<<(HID * G4  + 255) / 256, 256, 0, stream>>>(Whh_f, Whh_t, HID, G4);
  k_t16<<<(HID * HID + 255) / 256, 256, 0, stream>>>(Whg_f, Whg_t, HID, HID);
  k_t16<<<(DIN * HID + 255) / 256, 256, 0, stream>>>(Wmk_f, Wmk_t, DIN, HID);
  k_t16<<<(DEV * HID + 255) / 256, 256, 0, stream>>>(We_f,  We_t,  DEV, HID);
  k_t16<<<(HID * P1  + 255) / 256, 256, 0, stream>>>(Wp1_f, Wp1_t, HID, P1);

  // sequential recurrence: 8 independent workgroups (16 batch rows each)
  k_recur<<<BSZ / 16, 256, 0, stream>>>(x, events, td, Whh_t, Wpi_t, Whg_t,
                                        Wmk_t, We_t, bg, bhg, bmk, be, decay, feat);

  // fully-parallel predictor over all B*T rows (one 16-row tile per wave)
  k_pred<<<(BSZ * TSEQ / 16) / 8, 256, 0, stream>>>(feat, Wp1_t, bp1, Wp2, bp2, preds);

  // temporal first-difference output
  k_tmdo<<<((size_t)BSZ * TSEQ * DIN + 255) / 256, 256, 0, stream>>>(x, d1);
}
